// SAGE_28776280883867
// MI455X (gfx1250) — matrix operand order repaired and checked by reading, not compiled, not run on hardware
//
#include <hip/hip_runtime.h>

typedef __attribute__((ext_vector_type(16))) __bf16 v16bf;
typedef __attribute__((ext_vector_type(8)))  __bf16 v8bf;
typedef __attribute__((ext_vector_type(8)))  float  v8f;

#define NN 50000
#define NE 400000
#define KT512 16   // 512 / 32

// ---------------- utility: zero a float buffer (float4 granularity) ----------
__global__ void zero_f32(float* __restrict__ p, long n4) {
  long t = (long)blockIdx.x * blockDim.x + threadIdx.x;
  if (t < n4) ((float4*)p)[t] = make_float4(0.f, 0.f, 0.f, 0.f);
}

// ---------------- in-degree via f32 atomics ---------------------------------
__global__ void degree_kernel(const int* __restrict__ dst, float* __restrict__ deg, int nE) {
  int t = blockIdx.x * blockDim.x + threadIdx.x;
  if (t < nE) atomicAdd(&deg[dst[t]], 1.0f);
}

__global__ void recip_kernel(float* __restrict__ deg, int n) {
  int t = blockIdx.x * blockDim.x + threadIdx.x;
  if (t < n) deg[t] = 1.0f / (deg[t] + 1.0f);
}

// ---------------- edge aggregation: agg[dst] += h[src] -----------------------
// 128 threads per edge; float4 gather + 4 f32 atomics (L2-resident agg buffer)
__global__ void aggregate_kernel(const float* __restrict__ h,
                                 const int* __restrict__ src,
                                 const int* __restrict__ dst,
                                 float* __restrict__ agg, int nE) {
  long t = (long)blockIdx.x * blockDim.x + threadIdx.x;
  if (t >= (long)nE * 128) return;
  int e = (int)(t >> 7);
  int q = (int)(t & 127);
  int s = src[e], d = dst[e];
  float4 v = *(const float4*)(h + (size_t)s * 512 + q * 4);
  float* p = agg + (size_t)d * 512 + q * 4;
  atomicAdd(p + 0, v.x);
  atomicAdd(p + 1, v.y);
  atomicAdd(p + 2, v.z);
  atomicAdd(p + 3, v.w);
}

// ---------------- normalize + convert + pack into WMMA A-fragment layout -----
// A-matrix 16-bit 16x32 (MxK) per ISA 7.12.2:
//   lanes 0-15 : M=lane,   elems 0-7 -> K=0..7,  elems 8-15 -> K=16..23
//   lanes 16-31: M=lane-16, elems 0-7 -> K=8..15, elems 8-15 -> K=24..31
// Apack layout: [mTile][kTile][lane][16 elems] -> each lane loads 32B contiguous.
__global__ void prepare_kernel(const float* __restrict__ agg,
                               const float* __restrict__ h,
                               const float* __restrict__ recip,
                               __bf16* __restrict__ Apack, int nNodes) {
  int t = blockIdx.x * blockDim.x + threadIdx.x;
  if (t >= nNodes * 64) return;
  int m = t >> 6, k8 = t & 63;          // k8: which 8-wide K group
  float r = recip[m];
  const float4* ga = (const float4*)(agg + (size_t)m * 512 + k8 * 8);
  const float4* gh = (const float4*)(h   + (size_t)m * 512 + k8 * 8);
  float4 a0 = ga[0], a1 = ga[1];
  float4 h0 = gh[0], h1 = gh[1];
  v8bf o;
  o[0] = (__bf16)((a0.x + h0.x) * r);
  o[1] = (__bf16)((a0.y + h0.y) * r);
  o[2] = (__bf16)((a0.z + h0.z) * r);
  o[3] = (__bf16)((a0.w + h0.w) * r);
  o[4] = (__bf16)((a1.x + h1.x) * r);
  o[5] = (__bf16)((a1.y + h1.y) * r);
  o[6] = (__bf16)((a1.z + h1.z) * r);
  o[7] = (__bf16)((a1.w + h1.w) * r);
  int kTile = k8 >> 2, g = k8 & 3;                 // g: 8-block within 32-wide kTile
  int lane = (m & 15) | ((g & 1) << 4);            // odd 8-blocks live in upper lanes
  int e0 = (g & 2) ? 8 : 0;                        // blocks 2,3 -> elems 8..15
  size_t off = ((((size_t)(m >> 4)) * KT512 + kTile) * 32 + lane) * 16 + e0;
  *(v8bf*)(Apack + off) = o;
}

// ---------------- pack W (KxN fp32 row-major) into B-fragment layout ---------
// B-matrix 16-bit 32x16 (KxN): lanes 0-15 -> K=0..7 (elems 0-7), K=16..23 (elems 8-15), lanes
// 16-31 -> K=8..15, K=24..31. Bpack layout: [nTile][kTile][lane][16 elems].
__global__ void pack_w_kernel(const float* __restrict__ W, __bf16* __restrict__ Wp,
                              int KTl, int N) {
  int t = blockIdx.x * blockDim.x + threadIdx.x;
  if (t >= KTl * 32 * N) return;
  int e    = t & 15;
  int lane = (t >> 4) & 31;
  int rem  = t >> 9;
  int kTile = rem % KTl;
  int nTile = rem / KTl;
  int n = nTile * 16 + (lane & 15);
  int k = kTile * 32 + ((e >> 3) << 4) + ((lane >> 4) << 3) + (e & 7);
  Wp[t] = (__bf16)W[(size_t)k * N + n];
}

// ---------------- bf16 WMMA GEMM: out = relu(A @ W + b) ----------------------
// One wave per 16x64 output strip: 4 f32 accumulators, K loop in KTl (16) steps of 32.
__global__ void __launch_bounds__(256)
gemm_kernel(const __bf16* __restrict__ Apack,
            const __bf16* __restrict__ Bpack,
            const float* __restrict__ bias,
            float* __restrict__ out,
            int mTiles, int nStrips, int N, int KTl, int relu) {
  int gwave = (int)(((size_t)blockIdx.x * blockDim.x + threadIdx.x) >> 5);
  int lane  = threadIdx.x & 31;
  if (gwave >= mTiles * nStrips) return;   // wave-uniform: EXEC stays all-ones
  int mTile = gwave / nStrips;
  int nt0   = (gwave % nStrips) * 4;

  const __bf16* aP  = Apack + ((size_t)mTile * KTl * 32 + lane) * 16;
  const __bf16* bP0 = Bpack + ((size_t)(nt0 + 0) * KTl * 32 + lane) * 16;
  const __bf16* bP1 = Bpack + ((size_t)(nt0 + 1) * KTl * 32 + lane) * 16;
  const __bf16* bP2 = Bpack + ((size_t)(nt0 + 2) * KTl * 32 + lane) * 16;
  const __bf16* bP3 = Bpack + ((size_t)(nt0 + 3) * KTl * 32 + lane) * 16;

  v8f acc0 = {}, acc1 = {}, acc2 = {}, acc3 = {};
#pragma unroll 2
  for (int kt = 0; kt < KTl; ++kt) {
    v16bf a  = *(const v16bf*)(aP  + (size_t)kt * 512);
    v16bf b0 = *(const v16bf*)(bP0 + (size_t)kt * 512);
    v16bf b1 = *(const v16bf*)(bP1 + (size_t)kt * 512);
    v16bf b2 = *(const v16bf*)(bP2 + (size_t)kt * 512);
    v16bf b3 = *(const v16bf*)(bP3 + (size_t)kt * 512);
    acc0 = __builtin_amdgcn_wmma_f32_16x16x32_bf16(false, a, false, b0, (short)0, acc0, false, false);
    acc1 = __builtin_amdgcn_wmma_f32_16x16x32_bf16(false, a, false, b1, (short)0, acc1, false, false);
    acc2 = __builtin_amdgcn_wmma_f32_16x16x32_bf16(false, a, false, b2, (short)0, acc2, false, false);
    acc3 = __builtin_amdgcn_wmma_f32_16x16x32_bf16(false, a, false, b3, (short)0, acc3, false, false);
  }

  // C/D f32 16x16 layout: lane 0-15 -> N=lane, reg r -> M=r; lanes 16-31 -> M=r+8
  int mBase = mTile * 16 + ((lane >> 4) << 3);
  int nc = lane & 15;
  int n0 = (nt0 + 0) * 16 + nc, n1 = (nt0 + 1) * 16 + nc;
  int n2 = (nt0 + 2) * 16 + nc, n3 = (nt0 + 3) * 16 + nc;
  float bb0 = bias[n0], bb1 = bias[n1], bb2 = bias[n2], bb3 = bias[n3];
#pragma unroll
  for (int r = 0; r < 8; ++r) {
    float* o = out + (size_t)(mBase + r) * N;
    float x0 = acc0[r] + bb0, x1 = acc1[r] + bb1;
    float x2 = acc2[r] + bb2, x3 = acc3[r] + bb3;
    if (relu) { x0 = fmaxf(x0, 0.f); x1 = fmaxf(x1, 0.f);
                x2 = fmaxf(x2, 0.f); x3 = fmaxf(x3, 0.f); }
    o[n0] = x0; o[n1] = x1; o[n2] = x2; o[n3] = x3;
  }
}

extern "C" void kernel_launch(void* const* d_in, const int* in_sizes, int n_in,
                              void* d_out, int out_size, void* d_ws, size_t ws_size,
                              hipStream_t stream) {
  const float* feats = (const float*)d_in[0];
  const int*   src   = (const int*)d_in[1];
  const int*   dst   = (const int*)d_in[2];
  const float* W0 = (const float*)d_in[3];
  const float* b0 = (const float*)d_in[4];
  const float* W1 = (const float*)d_in[5];
  const float* b1 = (const float*)d_in[6];
  const float* W2 = (const float*)d_in[7];
  const float* b2 = (const float*)d_in[8];
  float* out = (float*)d_out;

  char* ws = (char*)d_ws;
  size_t off = 0;
  auto alloc = [&](size_t bytes) {
    char* p = ws + off;
    off = (off + bytes + 255) & ~(size_t)255;
    return p;
  };
  float*  deg   = (float*) alloc((size_t)NN * 4);
  float*  agg   = (float*) alloc((size_t)NN * 512 * 4);
  float*  h1    = (float*) alloc((size_t)NN * 512 * 4);
  float*  h2    = (float*) alloc((size_t)NN * 512 * 4);
  __bf16* Apack = (__bf16*)alloc((size_t)NN * 512 * 2);
  __bf16* Wp0   = (__bf16*)alloc((size_t)512 * 512 * 2);
  __bf16* Wp1   = (__bf16*)alloc((size_t)512 * 512 * 2);
  __bf16* Wp2   = (__bf16*)alloc((size_t)512 * 256 * 2);

  const int mTiles = NN / 16;               // 3125 (exact)
  const long aggN4 = (long)NN * 128;        // float4 count of agg
  const long aggT  = (long)NE * 128;        // aggregation threads

  // in-degrees -> 1/(deg+1)
  zero_f32<<<(NN / 4 + 255) / 256, 256, 0, stream>>>(deg, NN / 4);
  degree_kernel<<<(NE + 255) / 256, 256, 0, stream>>>(dst, deg, NE);
  recip_kernel<<<(NN + 255) / 256, 256, 0, stream>>>(deg, NN);

  // pack weights to bf16 B-fragment layout (once per launch)
  pack_w_kernel<<<(512 * 512 + 255) / 256, 256, 0, stream>>>(W0, Wp0, KT512, 512);
  pack_w_kernel<<<(512 * 512 + 255) / 256, 256, 0, stream>>>(W1, Wp1, KT512, 512);
  pack_w_kernel<<<(512 * 256 + 255) / 256, 256, 0, stream>>>(W2, Wp2, KT512, 256);

  // ---- Layer 0: feats -> h1 (relu) ----
  zero_f32<<<(int)((aggN4 + 255) / 256), 256, 0, stream>>>(agg, aggN4);
  aggregate_kernel<<<(int)((aggT + 255) / 256), 256, 0, stream>>>(feats, src, dst, agg, NE);
  prepare_kernel<<<(NN * 64 + 255) / 256, 256, 0, stream>>>(agg, feats, deg, Apack, NN);
  gemm_kernel<<<(mTiles * 8 * 32 + 255) / 256, 256, 0, stream>>>(
      Apack, Wp0, b0, h1, mTiles, 8, 512, KT512, 1);

  // ---- Layer 1: h1 -> h2 (relu) ----
  zero_f32<<<(int)((aggN4 + 255) / 256), 256, 0, stream>>>(agg, aggN4);
  aggregate_kernel<<<(int)((aggT + 255) / 256), 256, 0, stream>>>(h1, src, dst, agg, NE);
  prepare_kernel<<<(NN * 64 + 255) / 256, 256, 0, stream>>>(agg, h1, deg, Apack, NN);
  gemm_kernel<<<(mTiles * 8 * 32 + 255) / 256, 256, 0, stream>>>(
      Apack, Wp1, b1, h2, mTiles, 8, 512, KT512, 1);

  // ---- Layer 2: h2 -> out (no relu, N=256) ----
  zero_f32<<<(int)((aggN4 + 255) / 256), 256, 0, stream>>>(agg, aggN4);
  aggregate_kernel<<<(int)((aggT + 255) / 256), 256, 0, stream>>>(h2, src, dst, agg, NE);
  prepare_kernel<<<(NN * 64 + 255) / 256, 256, 0, stream>>>(agg, h2, deg, Apack, NN);
  gemm_kernel<<<(mTiles * 4 * 32 + 255) / 256, 256, 0, stream>>>(
      Apack, Wp2, b2, out, mTiles, 4, 256, KT512, 0);
}
